// LSTMDecoder_56418690400370
// MI455X (gfx1250) — matrix operand order repaired and checked by reading, not compiled, not run on hardware
//
#include <hip/hip_runtime.h>

#define HID   1024
#define BATCH 64
#define NLAYR 2
#define SEQT  128
#define G4    (4*HID)

typedef __attribute__((ext_vector_type(16))) __bf16 bf16x16;
typedef __attribute__((ext_vector_type(8)))  __bf16 bf16x8;
typedef __attribute__((ext_vector_type(8)))  float  f32x8;

__device__ __forceinline__ float sigmoidf_(float x) {
    return 1.0f / (1.0f + __expf(-x));
}

// ---- one-time setup kernels -------------------------------------------------
__global__ void cvt_f32_bf16(const float* __restrict__ src,
                             __bf16* __restrict__ dst, int n) {
    int i = blockIdx.x * blockDim.x + threadIdx.x;
    if (i < n) dst[i] = (__bf16)src[i];
}

__global__ void bias_sum(const float* __restrict__ bi,
                         const float* __restrict__ bh,
                         float* __restrict__ b, int n) {
    int i = blockIdx.x * blockDim.x + threadIdx.x;
    if (i < n) b[i] = bi[i] + bh[i];
}

__global__ void copy_f32(const float* __restrict__ src,
                         float* __restrict__ dst, int n) {
    int i = blockIdx.x * blockDim.x + threadIdx.x;
    if (i < n) dst[i] = src[i];
}

// ---- fused LSTM layer step --------------------------------------------------
// grid: 128 WGs = 2 batch-halves x 64 hidden-slices; block: 128 = 4 waves (one
// per gate). Each wave computes TWO 16x16 tiles (two batch tiles) against the
// SAME weight fragment, halving weight traffic and amortizing B loads:
// per K-step = 1 B frag + 2 A frags = 6 b128 loads for 2 WMMAs.
__global__ __launch_bounds__(128)
void lstm_layer_step(const __bf16* __restrict__ xin,    // [B,H] layer input (bf16)
                     const __bf16* __restrict__ hin,    // [B,H] this layer's h_{t-1}
                     const __bf16* __restrict__ Wih,    // [4H,H] bf16
                     const __bf16* __restrict__ Whh,    // [4H,H] bf16
                     const float*  __restrict__ bias,   // [4H] = b_ih + b_hh
                     float*        __restrict__ cst,    // [B,H] cell state (fp32, in/out)
                     __bf16*       __restrict__ hout,   // [B,H] h_t (bf16, next-step operand)
                     float*        __restrict__ yout)   // d_out + t*H (stride SEQT*H per batch), or null
{
    __shared__ float lds[4 * 2 * 256];   // [gate][local batch tile][16x16]

    const int gate = threadIdx.x >> 5;          // 0:i 1:f 2:g 3:o
    const int lane = threadIdx.x & 31;
    const int kh   = lane >> 4;                 // half-wave selector
    const int l16  = lane & 15;
    const int bh   = blockIdx.x & 1;            // batch half (rows bh*32 .. +31)
    const int hs   = blockIdx.x >> 1;           // hidden slice (16 cols)

    // B fragment (contiguous-K order, same as A): lane(kh,n) holds column n, K = 16*kh + [0..15] contiguous
    const int brow = gate * HID + hs * 16 + l16;       // weight row (gate output)
    const __bf16* bXp = Wih + (size_t)brow * HID + kh * 16;
    const __bf16* bHp = Whh + (size_t)brow * HID + kh * 16;

    // A fragments: lane(kh,m) holds K = 16*kh + [0..15] contiguous (matches B)
    const int arow0 = bh * 32 + l16;                   // batch tile 0 row
    const int arow1 = arow0 + 16;                      // batch tile 1 row
    const __bf16* aX0 = xin + (size_t)arow0 * HID + kh * 16;
    const __bf16* aX1 = xin + (size_t)arow1 * HID + kh * 16;
    const __bf16* aH0 = hin + (size_t)arow0 * HID + kh * 16;
    const __bf16* aH1 = hin + (size_t)arow1 * HID + kh * 16;

    f32x8 acc0 = {0.f, 0.f, 0.f, 0.f, 0.f, 0.f, 0.f, 0.f};
    f32x8 acc1 = {0.f, 0.f, 0.f, 0.f, 0.f, 0.f, 0.f, 0.f};

    union frag { bf16x16 v; bf16x8 h[2]; };

    #pragma unroll 4
    for (int k = 0; k < HID; k += 32) {
        bf16x16 b = *reinterpret_cast<const bf16x16*>(bXp + k);
        frag a0, a1;
        a0.h[0] = *reinterpret_cast<const bf16x8*>(aX0 + k);
        a0.h[1] = *reinterpret_cast<const bf16x8*>(aX0 + k + 8);
        a1.h[0] = *reinterpret_cast<const bf16x8*>(aX1 + k);
        a1.h[1] = *reinterpret_cast<const bf16x8*>(aX1 + k + 8);
        acc0 = __builtin_amdgcn_wmma_f32_16x16x32_bf16(
                   false, a0.v, false, b, (short)0, acc0, false, false);
        acc1 = __builtin_amdgcn_wmma_f32_16x16x32_bf16(
                   false, a1.v, false, b, (short)0, acc1, false, false);
    }
    #pragma unroll 4
    for (int k = 0; k < HID; k += 32) {
        bf16x16 b = *reinterpret_cast<const bf16x16*>(bHp + k);
        frag a0, a1;
        a0.h[0] = *reinterpret_cast<const bf16x8*>(aH0 + k);
        a0.h[1] = *reinterpret_cast<const bf16x8*>(aH0 + k + 8);
        a1.h[0] = *reinterpret_cast<const bf16x8*>(aH1 + k);
        a1.h[1] = *reinterpret_cast<const bf16x8*>(aH1 + k + 8);
        acc0 = __builtin_amdgcn_wmma_f32_16x16x32_bf16(
                   false, a0.v, false, b, (short)0, acc0, false, false);
        acc1 = __builtin_amdgcn_wmma_f32_16x16x32_bf16(
                   false, a1.v, false, b, (short)0, acc1, false, false);
    }

    // C/D layout: VGPR v, lanes 0-15 -> (M=v, N=lane); lanes 16-31 -> (M=v+8, N=lane-16)
    #pragma unroll
    for (int v = 0; v < 8; ++v) {
        lds[gate * 512 +       v * 32 + lane] = acc0[v];
        lds[gate * 512 + 256 + v * 32 + lane] = acc1[v];
    }
    __syncthreads();

    // 32 batch rows x 16 cols = 512 elements over 128 threads: fused cell update
    for (int e = threadIdx.x; e < 512; e += 128) {
        const int bl  = e >> 8;                   // local batch tile 0/1
        const int r   = e & 255;
        const int m   = r >> 4;
        const int n   = r & 15;
        const int idx = bl * 256 + (m & 7) * 32 + ((m >> 3) << 4) + n;  // invert C/D layout
        const int col = hs * 16 + n;
        const int bb  = bh * 32 + bl * 16 + m;

        float iv = lds[0 * 512 + idx] + bias[0 * HID + col];
        float fv = lds[1 * 512 + idx] + bias[1 * HID + col];
        float gv = lds[2 * 512 + idx] + bias[2 * HID + col];
        float ov = lds[3 * 512 + idx] + bias[3 * HID + col];

        const size_t cij = (size_t)bb * HID + col;
        float cp = cst[cij];
        float cn = sigmoidf_(fv) * cp + sigmoidf_(iv) * tanhf(gv);
        float hn = sigmoidf_(ov) * tanhf(cn);

        cst[cij]  = cn;
        hout[cij] = (__bf16)hn;
        if (yout) yout[(size_t)bb * (SEQT * HID) + col] = hn;  // yout already offset by t*H
    }
}

// ---- launch -----------------------------------------------------------------
extern "C" void kernel_launch(void* const* d_in, const int* in_sizes, int n_in,
                              void* d_out, int out_size, void* d_ws, size_t ws_size,
                              hipStream_t stream) {
    const float* x0  = (const float*)d_in[0];  // [B,1,H]
    const float* h0  = (const float*)d_in[1];  // [L,B,H]
    const float* c0  = (const float*)d_in[2];  // [L,B,H]
    const float* Wih = (const float*)d_in[3];  // [L,4H,H]
    const float* Whh = (const float*)d_in[4];  // [L,4H,H]
    const float* bih = (const float*)d_in[5];  // [L,4H]
    const float* bhh = (const float*)d_in[6];  // [L,4H]
    float* out = (float*)d_out;                // [B,T,H]

    // workspace carve-out (256B aligned slices)
    char*  ws  = (char*)d_ws;
    size_t off = 0;
    auto carve = [&](size_t bytes) -> void* {
        void* p = ws + off;
        off = (off + bytes + 255) & ~(size_t)255;
        return p;
    };
    const size_t nW = (size_t)NLAYR * G4 * HID;      // 8.39M elems per matrix set
    const size_t nS = (size_t)NLAYR * BATCH * HID;   // 131072
    __bf16* Wih_b = (__bf16*)carve(nW * sizeof(__bf16));
    __bf16* Whh_b = (__bf16*)carve(nW * sizeof(__bf16));
    float*  bsum  = (float*) carve((size_t)NLAYR * G4 * sizeof(float));
    float*  c_ws  = (float*) carve(nS * sizeof(float));
    __bf16* h_b   = (__bf16*)carve(nS * sizeof(__bf16));   // [L,B,H]
    __bf16* x0_b  = (__bf16*)carve((size_t)BATCH * HID * sizeof(__bf16));

    // one-time (per call) setup: bf16 weights, summed bias, state init
    {
        int n = (int)nW;
        cvt_f32_bf16<<<(n + 255) / 256, 256, 0, stream>>>(Wih, Wih_b, n);
        cvt_f32_bf16<<<(n + 255) / 256, 256, 0, stream>>>(Whh, Whh_b, n);
        int nb = NLAYR * G4;
        bias_sum<<<(nb + 255) / 256, 256, 0, stream>>>(bih, bhh, bsum, nb);
        int ns = (int)nS;
        cvt_f32_bf16<<<(ns + 255) / 256, 256, 0, stream>>>(h0, h_b, ns);
        copy_f32<<<(ns + 255) / 256, 256, 0, stream>>>(c0, c_ws, ns);
        int nx = BATCH * HID;
        cvt_f32_bf16<<<(nx + 255) / 256, 256, 0, stream>>>(x0, x0_b, nx);
    }

    const size_t SH = (size_t)BATCH * HID;   // per-layer state stride
    dim3 grid(128), block(128);

    for (int t = 0; t < SEQT; ++t) {
        // layer 0: input is x0 at t=0, else previous top-layer output (decoder feedback)
        const __bf16* in0 = (t == 0) ? x0_b : (h_b + SH);
        lstm_layer_step<<<grid, block, 0, stream>>>(
            in0, h_b,
            Wih_b, Whh_b, bsum,
            c_ws, h_b, nullptr);
        // layer 1: input is layer 0's fresh h; emit fp32 output row t
        lstm_layer_step<<<grid, block, 0, stream>>>(
            h_b, h_b + SH,
            Wih_b + (size_t)G4 * HID, Whh_b + (size_t)G4 * HID, bsum + G4,
            c_ws + SH, h_b + SH, out + (size_t)t * HID);
    }
}
